// BMETAdaptiveLogSoftmaxWithLoss_46694884442523
// MI455X (gfx1250) — matrix operand order repaired and checked by reading, not compiled, not run on hardware
//
#include <hip/hip_runtime.h>
#include <math.h>
#include <stdint.h>

// ---- problem constants (match reference) ----
#define NROWS   2048
#define KDIM    1024
#define CUT0    4000
#define CUT1    20000
#define NCLS    50257
#define HEADOUT 4002        // CUT0 + 2 cluster logits
#define H0DIM   512
#define H1DIM   256
#define T0OUT   (CUT1 - CUT0)   // 16000
#define T1OUT   (NCLS - CUT1)   // 30257
#define SMAX    8

typedef __attribute__((ext_vector_type(16))) __bf16   bf16x16;
typedef __attribute__((ext_vector_type(8)))  float    f32x8;
typedef __attribute__((ext_vector_type(4)))  uint32_t u32x4;

union FragU { bf16x16 v; u32x4 q[2]; };

__device__ __forceinline__ unsigned short f2bf(float f) {
  uint32_t u = __float_as_uint(f);
  return (unsigned short)((u + 0x7FFFu + ((u >> 16) & 1u)) >> 16);
}
__device__ __forceinline__ float bf2f(unsigned short h) {
  return __uint_as_float((uint32_t)h << 16);
}

// ---------------- f32 -> bf16 convert ----------------
__global__ void cvt_f32_bf16(const float* __restrict__ in,
                             unsigned short* __restrict__ out, int n) {
  int i = blockIdx.x * blockDim.x + threadIdx.x;
  if (i < n) out[i] = f2bf(in[i]);
}

// ---------------- per-row gather indices ----------------
__global__ void prep_idx(const long long* __restrict__ tgt,
                         int* __restrict__ head_col,
                         int* __restrict__ rel0, int* __restrict__ rel1, int n) {
  int i = blockIdx.x * blockDim.x + threadIdx.x;
  if (i >= n) return;
  long long t = tgt[i];
  head_col[i] = (t < CUT0) ? (int)t : ((t < CUT1) ? CUT0 : CUT0 + 1);
  int r0 = (int)t - CUT0;
  rel0[i] = r0 < 0 ? 0 : (r0 > T0OUT - 1 ? T0OUT - 1 : r0);
  int r1 = (int)t - CUT1;
  rel1[i] = r1 < 0 ? 0 : (r1 > T1OUT - 1 ? T1OUT - 1 : r1);
}

// ---------------- stage-1 tail GEMM: H = X @ W^T, bf16 out ----------------
template <int KD, int OUTD>
__global__ __launch_bounds__(256) void gemm_hidden(
    const unsigned short* __restrict__ X, const unsigned short* __restrict__ W,
    unsigned short* __restrict__ H) {
  __shared__ __attribute__((aligned(16))) unsigned short As[16 * KD];
  const int rb = blockIdx.x * 16;
  const u32x4* src = (const u32x4*)(X + (size_t)rb * KD);
  for (int i = threadIdx.x; i < 16 * KD / 8; i += 256)
    ((u32x4*)As)[i] = src[i];
  __syncthreads();

  const int lane = threadIdx.x & 31, wave = threadIdx.x >> 5;
  const int lh = lane >> 4, ln = lane & 15;
  for (int t = wave; t < OUTD / 16; t += 8) {
    const int col = t * 16 + ln;
    f32x8 acc = {0.f, 0.f, 0.f, 0.f, 0.f, 0.f, 0.f, 0.f};
    const unsigned short* wrow = W + (size_t)col * KD;
    for (int kk = 0; kk < KD; kk += 32) {
      FragU a, b;
      const unsigned short* ap = As + ln * KD + kk + lh * 8;
      a.q[0] = *(const u32x4*)ap;
      a.q[1] = *(const u32x4*)(ap + 16);
      const unsigned short* bp = wrow + kk + lh * 8;
      b.q[0] = *(const u32x4*)bp;
      b.q[1] = *(const u32x4*)(bp + 16);
      acc = __builtin_amdgcn_wmma_f32_16x16x32_bf16(false, a.v, false, b.v,
                                                    (short)0, acc, false, false);
    }
    for (int r = 0; r < 8; ++r)
      H[(size_t)(rb + r + lh * 8) * OUTD + col] = f2bf(acc[r]);
  }
}

// ------- fused GEMM + online partial log-sum-exp (column-split) -------
// Block: MT*16 rows x (1/S of columns). Emits per-(split,row) partial (max, sum).
template <int KD, int MT, int S>
__global__ __launch_bounds__(256) void fused_lse_part(
    const unsigned short* __restrict__ X,
    const unsigned short* __restrict__ W, int OUT,
    float* __restrict__ part_m, float* __restrict__ part_s) {
  __shared__ __attribute__((aligned(16))) unsigned short As[MT * 16 * KD];
  const int rb = blockIdx.x * (MT * 16);
  const int cs = blockIdx.y;
  const u32x4* src = (const u32x4*)(X + (size_t)rb * KD);
  for (int i = threadIdx.x; i < MT * 16 * KD / 8; i += 256)
    ((u32x4*)As)[i] = src[i];
  __syncthreads();

  const int lane = threadIdx.x & 31, wave = threadIdx.x >> 5;
  const int lh = lane >> 4, ln = lane & 15;
  float m[MT][8], s[MT][8];
  for (int mt = 0; mt < MT; ++mt)
    for (int r = 0; r < 8; ++r) { m[mt][r] = -INFINITY; s[mt][r] = 0.f; }

  const int ntiles = (OUT + 15) / 16;
  for (int t = cs * 8 + wave; t < ntiles; t += S * 8) {
    const int col = t * 16 + ln;
    const bool valid = col < OUT;
    const int ccol = valid ? col : OUT - 1;      // clamp: loads always in-bounds
    const unsigned short* wrow = W + (size_t)ccol * KD;
    if (col + S * 128 < OUT)
      __builtin_prefetch(W + (size_t)(col + S * 128) * KD, 0, 0);
    f32x8 acc[MT];
    for (int mt = 0; mt < MT; ++mt)
      acc[mt] = (f32x8){0.f, 0.f, 0.f, 0.f, 0.f, 0.f, 0.f, 0.f};
    for (int kk = 0; kk < KD; kk += 32) {
      FragU b;
      const unsigned short* bp = wrow + kk + lh * 8;
      b.q[0] = *(const u32x4*)bp;
      b.q[1] = *(const u32x4*)(bp + 16);
#pragma unroll
      for (int mt = 0; mt < MT; ++mt) {
        FragU a;
        const unsigned short* ap = As + (mt * 16 + ln) * KD + kk + lh * 8;
        a.q[0] = *(const u32x4*)ap;
        a.q[1] = *(const u32x4*)(ap + 16);
        acc[mt] = __builtin_amdgcn_wmma_f32_16x16x32_bf16(
            false, a.v, false, b.v, (short)0, acc[mt], false, false);
      }
    }
    if (valid) {
      for (int mt = 0; mt < MT; ++mt)
        for (int r = 0; r < 8; ++r) {
          float v = acc[mt][r];
          float nm = fmaxf(m[mt][r], v);
          s[mt][r] = s[mt][r] * __expf(m[mt][r] - nm) + __expf(v - nm);
          m[mt][r] = nm;
        }
    }
  }
  // butterfly across the 16 lanes of each half-wave (halves hold disjoint rows)
  for (int xm = 1; xm < 16; xm <<= 1) {
    for (int mt = 0; mt < MT; ++mt)
      for (int r = 0; r < 8; ++r) {
        float om = __shfl_xor(m[mt][r], xm, 32);
        float os = __shfl_xor(s[mt][r], xm, 32);
        float nm = fmaxf(m[mt][r], om);
        float e1 = (m[mt][r] == -INFINITY) ? 0.f : __expf(m[mt][r] - nm);
        float e2 = (om == -INFINITY) ? 0.f : __expf(om - nm);
        s[mt][r] = s[mt][r] * e1 + os * e2;
        m[mt][r] = nm;
      }
  }
  // reuse As as cross-wave reduction scratch (all waves done reading it)
  __syncthreads();
  float* redm = (float*)As;                  // [8][MT*16]
  float* reds = redm + 8 * MT * 16;          // [8][MT*16]
  if (ln == 0)
    for (int mt = 0; mt < MT; ++mt)
      for (int r = 0; r < 8; ++r) {
        redm[wave * (MT * 16) + mt * 16 + r + lh * 8] = m[mt][r];
        reds[wave * (MT * 16) + mt * 16 + r + lh * 8] = s[mt][r];
      }
  __syncthreads();
  if (threadIdx.x < MT * 16) {
    const int r = threadIdx.x;
    float fm = -INFINITY, fs = 0.f;
    for (int w = 0; w < 8; ++w) fm = fmaxf(fm, redm[w * (MT * 16) + r]);
    for (int w = 0; w < 8; ++w) {
      float mm = redm[w * (MT * 16) + r];
      if (mm != -INFINITY) fs += reds[w * (MT * 16) + r] * __expf(mm - fm);
    }
    part_m[(size_t)cs * NROWS + rb + r] = fm;
    part_s[(size_t)cs * NROWS + rb + r] = fs;
  }
}

// ---------------- merge column-split partials -> lse ----------------
__global__ void combine_lse(const float* __restrict__ pm,
                            const float* __restrict__ ps, int S,
                            float* __restrict__ lse, int n) {
  int i = blockIdx.x * blockDim.x + threadIdx.x;
  if (i >= n) return;
  float fm = -INFINITY;
  for (int s = 0; s < S; ++s) fm = fmaxf(fm, pm[s * n + i]);
  float fs = 0.f;
  for (int s = 0; s < S; ++s) {
    float mm = pm[s * n + i];
    if (mm != -INFINITY) fs += ps[s * n + i] * __expf(mm - fm);
  }
  lse[i] = fm + __logf(fs);
}

// ---------------- target logit = dot(X[row], W[tcol[row]]) ----------------
__global__ __launch_bounds__(256) void gather_dot(
    const unsigned short* __restrict__ X, int K,
    const unsigned short* __restrict__ W,
    const int* __restrict__ tcol, float* __restrict__ out, int nrows) {
  const int wave = threadIdx.x >> 5, lane = threadIdx.x & 31;
  const int row = blockIdx.x * 8 + wave;
  if (row >= nrows) return;
  const int tc = tcol[row];
  const unsigned short* xr = X + (size_t)row * K;
  const unsigned short* wr = W + (size_t)tc * K;
  float acc = 0.f;
  for (int k = lane; k < K; k += 32)
    acc = fmaf(bf2f(xr[k]), bf2f(wr[k]), acc);
  for (int xm = 16; xm >= 1; xm >>= 1) acc += __shfl_xor(acc, xm, 32);
  if (lane == 0) out[row] = acc;
}

// ---------------- combine terms -> per-sample NLL ----------------
__global__ void finalize(const long long* __restrict__ tgt,
                         const float* __restrict__ h_lse, const float* __restrict__ h_tgt,
                         const float* __restrict__ l0_lse, const float* __restrict__ l0_tgt,
                         const float* __restrict__ l1_lse, const float* __restrict__ l1_tgt,
                         float* __restrict__ out, int n) {
  int i = blockIdx.x * blockDim.x + threadIdx.x;
  if (i >= n) return;
  long long t = tgt[i];
  float v = h_tgt[i] - h_lse[i];
  if (t >= CUT0 && t < CUT1) v += l0_tgt[i] - l0_lse[i];
  if (t >= CUT1)             v += l1_tgt[i] - l1_lse[i];
  out[i] = -v;
}

extern "C" void kernel_launch(void* const* d_in, const int* in_sizes, int n_in,
                              void* d_out, int out_size, void* d_ws, size_t ws_size,
                              hipStream_t stream) {
  const float*     inp   = (const float*)d_in[0];
  const long long* tgt   = (const long long*)d_in[1];
  const float*     headw = (const float*)d_in[2];
  const float*     t0w1  = (const float*)d_in[3];
  const float*     t0w2  = (const float*)d_in[4];
  const float*     t1w1  = (const float*)d_in[5];
  const float*     t1w2  = (const float*)d_in[6];
  float* out = (float*)d_out;

  char* wsp = (char*)d_ws;
  auto alloc = [&](size_t bytes) -> char* {
    char* p = wsp; wsp += (bytes + 255) & ~(size_t)255; return p;
  };
  unsigned short* inp_b   = (unsigned short*)alloc((size_t)NROWS * KDIM * 2);
  unsigned short* headw_b = (unsigned short*)alloc((size_t)HEADOUT * KDIM * 2);
  unsigned short* t0w1_b  = (unsigned short*)alloc((size_t)H0DIM * KDIM * 2);
  unsigned short* t0w2_b  = (unsigned short*)alloc((size_t)T0OUT * H0DIM * 2);
  unsigned short* t1w1_b  = (unsigned short*)alloc((size_t)H1DIM * KDIM * 2);
  unsigned short* t1w2_b  = (unsigned short*)alloc((size_t)T1OUT * H1DIM * 2);
  unsigned short* h0_b    = (unsigned short*)alloc((size_t)NROWS * H0DIM * 2);
  unsigned short* h1_b    = (unsigned short*)alloc((size_t)NROWS * H1DIM * 2);
  float* pm     = (float*)alloc((size_t)SMAX * NROWS * 4);
  float* ps     = (float*)alloc((size_t)SMAX * NROWS * 4);
  float* h_lse  = (float*)alloc(NROWS * 4);
  float* h_tgt  = (float*)alloc(NROWS * 4);
  float* l0_lse = (float*)alloc(NROWS * 4);
  float* l0_tgt = (float*)alloc(NROWS * 4);
  float* l1_lse = (float*)alloc(NROWS * 4);
  float* l1_tgt = (float*)alloc(NROWS * 4);
  int* head_col = (int*)alloc(NROWS * 4);
  int* rel0     = (int*)alloc(NROWS * 4);
  int* rel1     = (int*)alloc(NROWS * 4);

  auto cvt = [&](const float* src, unsigned short* dst, size_t n) {
    cvt_f32_bf16<<<(unsigned)((n + 255) / 256), 256, 0, stream>>>(src, dst, (int)n);
  };
  cvt(inp,   inp_b,   (size_t)NROWS * KDIM);
  cvt(headw, headw_b, (size_t)HEADOUT * KDIM);
  cvt(t0w1,  t0w1_b,  (size_t)H0DIM * KDIM);
  cvt(t0w2,  t0w2_b,  (size_t)T0OUT * H0DIM);
  cvt(t1w1,  t1w1_b,  (size_t)H1DIM * KDIM);
  cvt(t1w2,  t1w2_b,  (size_t)T1OUT * H1DIM);

  prep_idx<<<NROWS / 256, 256, 0, stream>>>(tgt, head_col, rel0, rel1, NROWS);

  gemm_hidden<KDIM, H0DIM><<<NROWS / 16, 256, 0, stream>>>(inp_b, t0w1_b, h0_b);
  gemm_hidden<KDIM, H1DIM><<<NROWS / 16, 256, 0, stream>>>(inp_b, t1w1_b, h1_b);

  // head: 32 rows/block (64KB A-tile), 4 column splits -> 256 blocks
  fused_lse_part<KDIM, 2, 4><<<dim3(NROWS / 32, 4), 256, 0, stream>>>(
      inp_b, headw_b, HEADOUT, pm, ps);
  combine_lse<<<NROWS / 256, 256, 0, stream>>>(pm, ps, 4, h_lse, NROWS);
  gather_dot<<<NROWS / 8, 256, 0, stream>>>(inp_b, KDIM, headw_b, head_col,
                                            h_tgt, NROWS);

  // tail0: 64 rows/block (64KB A-tile), 8 splits -> 256 blocks
  fused_lse_part<H0DIM, 4, 8><<<dim3(NROWS / 64, 8), 256, 0, stream>>>(
      h0_b, t0w2_b, T0OUT, pm, ps);
  combine_lse<<<NROWS / 256, 256, 0, stream>>>(pm, ps, 8, l0_lse, NROWS);
  gather_dot<<<NROWS / 8, 256, 0, stream>>>(h0_b, H0DIM, t0w2_b, rel0,
                                            l0_tgt, NROWS);

  // tail1: 64 rows/block (32KB A-tile), 8 splits -> 256 blocks
  fused_lse_part<H1DIM, 4, 8><<<dim3(NROWS / 64, 8), 256, 0, stream>>>(
      h1_b, t1w2_b, T1OUT, pm, ps);
  combine_lse<<<NROWS / 256, 256, 0, stream>>>(pm, ps, 8, l1_lse, NROWS);
  gather_dot<<<NROWS / 8, 256, 0, stream>>>(h1_b, H1DIM, t1w2_b, rel1,
                                            l1_tgt, NROWS);

  finalize<<<NROWS / 256, 256, 0, stream>>>(tgt, h_lse, h_tgt, l0_lse, l0_tgt,
                                            l1_lse, l1_tgt, out, NROWS);
}
